// SimpleRNN_45217415693034
// MI455X (gfx1250) — matrix operand order repaired and checked by reading, not compiled, not run on hardware
//
#include <hip/hip_runtime.h>
#include <hip/hip_bf16.h>
#include <stdint.h>

// Problem constants
#define V_ 8192
#define H_ 1024
#define B_ 64
#define S_ 256

typedef _Float16 v16h __attribute__((ext_vector_type(16)));
typedef _Float16 v8h  __attribute__((ext_vector_type(8)));
typedef float    v8f  __attribute__((ext_vector_type(8)));
typedef unsigned int u32x4 __attribute__((ext_vector_type(4)));
typedef int i32x8 __attribute__((ext_vector_type(8)));
typedef int i32x4 __attribute__((ext_vector_type(4)));

__device__ __forceinline__ v8f wmma16(v16h a, v16h b, v8f c) {
  // v_wmma_f32_16x16x32_f16 : D = A(16x32 f16) * B(32x16 f16) + C(16x16 f32)
  return __builtin_amdgcn_wmma_f32_16x16x32_f16(false, a, false, b, (short)0, c,
                                                false, false);
}

// ---------------------------------------------------------------------------
// Utility kernels
// ---------------------------------------------------------------------------
__global__ __launch_bounds__(256) void k_f32_to_f16(const float* __restrict__ src,
                                                    _Float16* __restrict__ dst, int n) {
  int i = blockIdx.x * 256 + threadIdx.x;
  if (i < n) dst[i] = (_Float16)src[i];
}

// Transpose W_ih (H,V) row-major -> W_ihT (V,H) row-major, fp32
__global__ __launch_bounds__(256) void k_transpose(const float* __restrict__ src,
                                                   float* __restrict__ dst) {
  __shared__ float tile[32][33];
  int v = blockIdx.x * 32 + threadIdx.x;   // column in src
  int h = blockIdx.y * 32 + threadIdx.y;   // row in src
  for (int j = 0; j < 32; j += 8)
    tile[threadIdx.y + j][threadIdx.x] = src[(size_t)(h + j) * V_ + v];
  __syncthreads();
  int vo = blockIdx.x * 32 + threadIdx.y;
  int ho = blockIdx.y * 32 + threadIdx.x;
  for (int j = 0; j < 32; j += 8)
    dst[(size_t)(vo + j) * H_ + ho] = tile[threadIdx.x][threadIdx.y + j];
}

// emb[s,b,:] = W_ihT[x[b,s],:] + b_ih + b_hh   (fp32, coalesced float4)
__global__ __launch_bounds__(256) void k_embed(const int* __restrict__ x,
                                               const float* __restrict__ wihT,
                                               const float* __restrict__ bih,
                                               const float* __restrict__ bhh,
                                               float* __restrict__ emb) {
  int bid = blockIdx.x;            // = s*B + b
  int s = bid >> 6, b = bid & 63;
  int idx = x[b * S_ + s];         // x is (B,S)
  int h4 = threadIdx.x * 4;
  float4 w = *(const float4*)(wihT + (size_t)idx * H_ + h4);
  float4 p = *(const float4*)(bih + h4);
  float4 q = *(const float4*)(bhh + h4);
  float4 r;
  r.x = w.x + p.x + q.x; r.y = w.y + p.y + q.y;
  r.z = w.z + p.z + q.z; r.w = w.w + p.w + q.w;
  *(float4*)(emb + (size_t)bid * H_ + h4) = r;
}

// ---------------------------------------------------------------------------
// Grid barrier (sense-reversing, agent scope) for the persistent scan kernel
// ---------------------------------------------------------------------------
__device__ __forceinline__ void grid_barrier(unsigned* bar, unsigned nwg) {
  __syncthreads();
  if (threadIdx.x == 0) {
    __threadfence();
    unsigned* cnt = bar;
    unsigned* gen = bar + 1;
    unsigned g = __hip_atomic_load(gen, __ATOMIC_ACQUIRE, __HIP_MEMORY_SCOPE_AGENT);
    unsigned a = __hip_atomic_fetch_add(cnt, 1u, __ATOMIC_ACQ_REL, __HIP_MEMORY_SCOPE_AGENT);
    if (a + 1 == nwg) {
      __hip_atomic_store(cnt, 0u, __ATOMIC_RELAXED, __HIP_MEMORY_SCOPE_AGENT);
      __hip_atomic_fetch_add(gen, 1u, __ATOMIC_RELEASE, __HIP_MEMORY_SCOPE_AGENT);
    } else {
      while (__hip_atomic_load(gen, __ATOMIC_ACQUIRE, __HIP_MEMORY_SCOPE_AGENT) == g) {
        __builtin_amdgcn_s_sleep(1);
      }
    }
  }
  __syncthreads();
}

// ---------------------------------------------------------------------------
// Persistent Elman scan: h_t = tanh(emb_t + h_{t-1} @ W_hh^T)
// 16 WGs x 256 threads. WG wg owns output columns [wg*64, wg*64+64).
// Its 64x1024 f16 slice of W_hh lives in LDS for all 256 steps.
// hbuf holds TWO adjacent (B,H) f16 state buffers (parity double buffer) so
// that all h accesses derive from one kernel arg -> global_load, not flat.
// ---------------------------------------------------------------------------
#define SCAN_ROWH 1032   // halves per LDS row: 1024 + 8 pad (2064B, bank-spread)

__global__ __launch_bounds__(256) void k_scan(const _Float16* __restrict__ whh16,
                                              const float* __restrict__ emb,
                                              _Float16* __restrict__ hbuf,
                                              _Float16* __restrict__ y16,
                                              float* __restrict__ hlast,
                                              unsigned* __restrict__ bar) {
  extern __shared__ _Float16 lds[];            // 64 * SCAN_ROWH halves = 129KB
  const int c0 = blockIdx.x * 64;              // first output column of this WG

  // Preload W_hh rows c0..c0+63 (B-operand needs W_hh[col, k] contiguous in k)
  for (int i = threadIdx.x; i < 64 * 128; i += 256) {
    int r = i >> 7, cv = i & 127;
    *(v8h*)(lds + r * SCAN_ROWH + cv * 8) =
        *(const v8h*)(whh16 + (size_t)(c0 + r) * H_ + cv * 8);
  }
  __syncthreads();

  const int wave = threadIdx.x >> 5, lane = threadIdx.x & 31;
  const int mt = wave & 3;                     // M tile (4 x 16 = 64 batch rows)
  const int nb = (wave >> 2) * 2;              // first of 2 N tiles
  const int lr = lane & 15, hi = lane >> 4;

  const int arow = mt * 16 + lr;
  const _Float16* b0base = lds + (nb * 16 + lr) * SCAN_ROWH + hi * 16;
  const _Float16* b1base = b0base + 16 * SCAN_ROWH;

  const size_t HBUF = (size_t)B_ * H_;         // elements per state buffer

  for (int s = 0; s < S_; ++s) {
    const size_t poff = (size_t)(s & 1) * HBUF;
    const size_t noff = (size_t)((s + 1) & 1) * HBUF;
    const _Float16* aptr = hbuf + poff + (size_t)arow * H_ + hi * 16;
    _Float16* hn = hbuf + noff;

    v8f acc0 = {}; v8f acc1 = {};
    for (int k = 0; k < H_; k += 32) {
      union { v16h v; v8h h[2]; } A, B0, B1;
      A.h[0] = *(const v8h*)(aptr + k);
      A.h[1] = *(const v8h*)(aptr + k + 8);
      B0.h[0] = *(const v8h*)(b0base + k);
      B0.h[1] = *(const v8h*)(b0base + k + 8);
      B1.h[0] = *(const v8h*)(b1base + k);
      B1.h[1] = *(const v8h*)(b1base + k + 8);
      acc0 = wmma16(A.v, B0.v, acc0);
      acc1 = wmma16(A.v, B1.v, acc1);
    }

    // Epilogue: add emb, tanh, emit h (f16), y (f16), final state (f32)
    const int col0 = c0 + nb * 16 + lr;
    const int col1 = col0 + 16;
    for (int r = 0; r < 8; ++r) {
      int row = mt * 16 + r + hi * 8;          // batch index
      size_t sb = (size_t)s * B_ + row;
      float e0 = emb[sb * H_ + col0];
      float e1 = emb[sb * H_ + col1];
      float v0 = tanhf(acc0[r] + e0);
      float v1 = tanhf(acc1[r] + e1);
      hn[(size_t)row * H_ + col0] = (_Float16)v0;
      hn[(size_t)row * H_ + col1] = (_Float16)v1;
      y16[sb * H_ + col0] = (_Float16)v0;
      y16[sb * H_ + col1] = (_Float16)v1;
      if (s == S_ - 1) {
        hlast[(size_t)row * H_ + col0] = v0;
        hlast[(size_t)row * H_ + col1] = v1;
      }
    }
    grid_barrier(bar, gridDim.x);
  }
}

// ---------------------------------------------------------------------------
// Output projection: ans[m, v] = sum_h y[m,h]*fcW[v,h] + fc_b[v]
// M = S*B = 16384, N = V = 8192, K = 1024. 64x64 WG tiles, f16 WMMA.
// fc_W tiles are DMA'd into LDS by the Tensor Data Mover with double buffering.
// ---------------------------------------------------------------------------
#define GEMM_ROWH 40     // halves per LDS tile row: 32 data + 4-DWORD TDM pad (80B)

__device__ __forceinline__ void tdm_load_tile(uint32_t lds_addr, uint64_t gaddr) {
  // D# group 0: count=1, lds_addr, 57-bit global addr, type=2 (image)
  u32x4 g0 = { 1u, lds_addr, (uint32_t)gaddr,
               (uint32_t)((gaddr >> 32) & 0x01FFFFFFu) | (2u << 30) };
  // D# group 1: data_size=2B(code1), pad_enable, pad_interval=16DW(code3),
  //             pad_amount=4DW(code3); tensor 1024x8192; tile 32x64; stride0=1024
  i32x8 g1 = { (int)((1u << 16) | (1u << 20) | (3u << 22) | (3u << 25)),
               (int)(1024u << 16),            // tensor_dim0[15:0] at [31:16]
               (int)(8192u << 16),            // td0 hi=0 | tensor_dim1[15:0]
               (int)(32u << 16),              // td1 hi=0 | tile_dim0=32
               (int)64,                       // tile_dim1=64, tile_dim2=0
               (int)1024,                     // tensor_dim0_stride low
               0, 0 };
  i32x4 z4 = {0, 0, 0, 0};
  i32x8 z8 = {0, 0, 0, 0, 0, 0, 0, 0};
  // 6-arg form (clang-23 / therock-10.0 HIP headers)
  __builtin_amdgcn_tensor_load_to_lds(g0, g1, z4, z4, z8, 0);
}

__global__ __launch_bounds__(256) void k_out_gemm(const _Float16* __restrict__ y16,
                                                  const _Float16* __restrict__ fcw16,
                                                  const float* __restrict__ fcb,
                                                  float* __restrict__ out) {
  __shared__ _Float16 tile[2][64 * GEMM_ROWH];   // 2 x 5120B

  const int v0 = blockIdx.x * 64;
  const int m0 = blockIdx.y * 64;
  const int wave = threadIdx.x >> 5, lane = threadIdx.x & 31;
  const int mt = wave & 3;
  const int nb = (wave >> 2) * 2;
  const int lr = lane & 15, hi = lane >> 4;

  const uint64_t gbase = (uint64_t)(uintptr_t)fcw16 + ((uint64_t)v0 * H_) * 2u;
  const uint32_t ldsb[2] = { (uint32_t)(uintptr_t)&tile[0][0],
                             (uint32_t)(uintptr_t)&tile[1][0] };

  if (wave == 0) tdm_load_tile(ldsb[0], gbase);          // prologue: kb=0 tile

  const _Float16* aptr = y16 + (size_t)(m0 + mt * 16 + lr) * H_ + hi * 16;
  v8f acc0 = {}; v8f acc1 = {};

  const int NKB = H_ / 32;   // 32
  for (int kb = 0; kb < NKB; ++kb) {
    __syncthreads();   // everyone finished reading the buffer we overwrite next
    if (wave == 0) {
      if (kb + 1 < NKB) {
        tdm_load_tile(ldsb[(kb + 1) & 1], gbase + (uint64_t)(kb + 1) * 32u * 2u);
        __builtin_amdgcn_s_wait_tensorcnt(1);  // current tile done, next in flight
      } else {
        __builtin_amdgcn_s_wait_tensorcnt(0);
      }
    }
    __syncthreads();   // tile[kb&1] visible to all waves

    const _Float16* bbase = &tile[kb & 1][0];
    const _Float16* b0p = bbase + (nb * 16 + lr) * GEMM_ROWH + hi * 16;
    const _Float16* b1p = b0p + 16 * GEMM_ROWH;
    const int k = kb * 32;

    union { v16h v; v8h h[2]; } A, B0, B1;
    A.h[0] = *(const v8h*)(aptr + k);
    A.h[1] = *(const v8h*)(aptr + k + 8);
    B0.h[0] = *(const v8h*)(b0p);
    B0.h[1] = *(const v8h*)(b0p + 8);
    B1.h[0] = *(const v8h*)(b1p);
    B1.h[1] = *(const v8h*)(b1p + 8);
    acc0 = wmma16(A.v, B0.v, acc0);
    acc1 = wmma16(A.v, B1.v, acc1);
  }

  const int gv0 = v0 + nb * 16 + lr;
  const int gv1 = gv0 + 16;
  const float bias0 = fcb[gv0];
  const float bias1 = fcb[gv1];
  for (int r = 0; r < 8; ++r) {
    int rowM = m0 + mt * 16 + r + hi * 8;
    out[(size_t)rowM * V_ + gv0] = acc0[r] + bias0;
    out[(size_t)rowM * V_ + gv1] = acc1[r] + bias1;
  }
}

// ---------------------------------------------------------------------------
// Host-side launch
// ---------------------------------------------------------------------------
extern "C" void kernel_launch(void* const* d_in, const int* in_sizes, int n_in,
                              void* d_out, int out_size, void* d_ws, size_t ws_size,
                              hipStream_t stream) {
  (void)in_sizes; (void)n_in; (void)out_size; (void)ws_size;

  const int*   x    = (const int*)d_in[0];    // (B,S) int32
  const float* st   = (const float*)d_in[1];  // (1,B,H)
  const float* Wih  = (const float*)d_in[2];  // (H,V)
  const float* Whh  = (const float*)d_in[3];  // (H,H)
  const float* bih  = (const float*)d_in[4];  // (H,)
  const float* bhh  = (const float*)d_in[5];  // (H,)
  const float* fcW  = (const float*)d_in[6];  // (V,H)
  const float* fcb  = (const float*)d_in[7];  // (V,)

  float* out = (float*)d_out;                          // ans (S,B,V)
  float* hlast = out + (size_t)S_ * B_ * V_;           // then state (1,B,H)

  char* ws = (char*)d_ws;
  size_t off = 0;
  float*    wihT  = (float*)(ws + off);    off += (size_t)V_ * H_ * 4;   // 32MB
  float*    emb   = (float*)(ws + off);    off += (size_t)S_ * B_ * H_ * 4; // 64MB
  _Float16* whh16 = (_Float16*)(ws + off); off += (size_t)H_ * H_ * 2;   // 2MB
  _Float16* fcw16 = (_Float16*)(ws + off); off += (size_t)V_ * H_ * 2;   // 16MB
  _Float16* y16   = (_Float16*)(ws + off); off += (size_t)S_ * B_ * H_ * 2; // 32MB
  _Float16* hbuf  = (_Float16*)(ws + off); off += 2 * (size_t)B_ * H_ * 2; // parity pair
  unsigned* bar   = (unsigned*)(ws + off); off += 256;

  // 1) Precision conversion of the two GEMM weight operands + initial state
  k_f32_to_f16<<<(H_ * H_) / 256, 256, 0, stream>>>(Whh, whh16, H_ * H_);
  k_f32_to_f16<<<(V_ * H_) / 256, 256, 0, stream>>>(fcW, fcw16, V_ * H_);
  k_f32_to_f16<<<(B_ * H_) / 256, 256, 0, stream>>>(st, hbuf, B_ * H_);

  // 2) Transpose W_ih so the embedding gather is coalesced
  k_transpose<<<dim3(V_ / 32, H_ / 32), dim3(32, 8), 0, stream>>>(Wih, wihT);

  // 3) Embedding gather + fused biases -> emb (S,B,H) fp32
  k_embed<<<S_ * B_, 256, 0, stream>>>(x, wihT, bih, bhh, emb);

  // 4) Barrier state must be zero each call (ws is not re-poisoned)
  (void)hipMemsetAsync(bar, 0, 64, stream);

  // 5) Persistent recurrence scan: 16 WGs, W_hh slice resident in 129KB LDS
  k_scan<<<16, 256, 64 * SCAN_ROWH * sizeof(_Float16), stream>>>(
      whh16, emb, hbuf, y16, hlast, bar);

  // 6) Output projection (S*B x V), TDM-fed double-buffered WMMA GEMM
  k_out_gemm<<<dim3(V_ / 64, (S_ * B_) / 64), 256, 0, stream>>>(y16, fcw16, fcb, out);
}
